// SD_67886253080719
// MI455X (gfx1250) — compile-verified, matrix operand order checked by reading, not run on hardware
//
#include <hip/hip_runtime.h>
#include <math.h>

// ---------------------------------------------------------------------------
// SA-DDI style GNN forward for MI455X (gfx1250, wave32, WMMA).
// All dense GEMMs run through a WMMA f16-in/f32-acc tile kernel
// (v_wmma_f32_16x16x32_f16) templated on K so masks constant-fold and the
// K-loop fully unrolls (branch-free fragment loads, float4 A loads,
// LDS-staged B shared by the 4 waves of a block).
// Graph-local ops (edge-softmax attention, graph-LayerNorm, SAG pooling,
// tiny 4-token co-attention) are one workgroup per graph using LDS.
//
// Workspace requirement: ~296 MiB (documented offsets below).
// ---------------------------------------------------------------------------

#define BGR   2048      // graphs per drug
#define NPGc  32        // nodes per graph
#define EPGc  64        // edges per graph
#define NTOT  (BGR*NPGc)   // 65536
#define ETOT  (BGR*EPGc)   // 131072
#define FINc  70
#define EINc  6
#define HIDc  128
#define NBLKc 4
#define CAHc  8
#define CADc  128

typedef __attribute__((ext_vector_type(16))) _Float16 v16h;
typedef __attribute__((ext_vector_type(8)))  float    v8f;

// Branch-free ELU: single hardware v_exp_f32 (TRANS32) + cndmask, no exec
// branches (expm1f would expand to a ~25-op polynomial under saveexec).
__device__ __forceinline__ float eluf(float x) {
    const float e = __expf(fminf(x, 0.f)) - 1.f;
    return x > 0.f ? x : e;
}

// ---------------------------------------------------------------------------
// WMMA GEMM:  C[M,Nout] = act( A[M,K] @ W[K,Nout] + bias )
// One wave computes one 16x16 tile of C; 4 waves / block share col tile.
// K compile-time; M % 64 == 0, Nout % 16 == 0.
// A-fragment layout per CDNA5 ISA 16-bit A-matrix (16x32):
//   lane L: row M = L%16, elements 0..7 -> K = k0+8*hi+j, 8..15 -> +16,
//   hi = L>=16. B-fragment mirrored (col N = L%16, same K striping),
//   fed from an LDS-staged 32x16 tile shared by all 4 waves.
// D layout: VGPR j -> row = row0 + j + 8*hi, col = col0 + L%16.
// ---------------------------------------------------------------------------
template <int K>
__global__ void __launch_bounds__(128)
k_gemm(const float* __restrict__ A, const float* __restrict__ W,
       const float* __restrict__ bias, float* __restrict__ C,
       int Nout, int act)
{
    constexpr int BSTR = 33;                 // padded LDS stride (bank-safe)
    __shared__ float Bs[16 * BSTR];          // [n][kk] tile, kk = K - k0

    const int t    = threadIdx.x;
    const int lane = t & 31;
    const int wave = t >> 5;
    const int row0 = (blockIdx.x * 4 + wave) * 16;
    const int col0 = blockIdx.y * 16;
    const int nn   = lane & 15;
    const int hi   = lane >> 4;
    const int m    = row0 + nn;              // A row this lane supplies
    const int n    = col0 + nn;              // C col this lane holds
    const float* __restrict__ Arow = A + (size_t)m * K;

    v8f acc = {};
#pragma unroll
    for (int k0 = 0; k0 < K; k0 += 32) {
        if (k0 + 32 < K)                     // global_prefetch_b8 next A tile
            __builtin_prefetch(Arow + k0 + 32, 0, 1);

        // ---- cooperative B tile stage: 32(K) x 16(N), coalesced rows ----
#pragma unroll
        for (int r = 0; r < 4; ++r) {
            const int idx = t + 128 * r;     // 512 elements
            const int nc  = idx & 15;
            const int kk  = idx >> 4;        // 0..31
            const int kg  = k0 + kk;
            if constexpr (K % 32 == 0) {
                Bs[nc * BSTR + kk] = W[(size_t)kg * Nout + col0 + nc];
            } else {
                const int kgc = kg < K ? kg : 0;
                const float w = W[(size_t)kgc * Nout + col0 + nc];
                Bs[nc * BSTR + kk] = (kg < K) ? w : 0.f;
            }
        }
        __syncthreads();

        // ---- fragments ----
        v16h af, bf;
        const int kaBase = k0 + 8 * hi;
        if constexpr (K % 8 == 0) {
            // 32B-aligned contiguous runs -> global_load_b128 pairs
            const float4* p0 = reinterpret_cast<const float4*>(Arow + kaBase);
            const float4* p1 = reinterpret_cast<const float4*>(Arow + kaBase + 16);
            const float4 a0 = p0[0], a1 = p0[1], a2 = p1[0], a3 = p1[1];
            af[0] = (_Float16)a0.x; af[1] = (_Float16)a0.y;
            af[2] = (_Float16)a0.z; af[3] = (_Float16)a0.w;
            af[4] = (_Float16)a1.x; af[5] = (_Float16)a1.y;
            af[6] = (_Float16)a1.z; af[7] = (_Float16)a1.w;
            af[8]  = (_Float16)a2.x; af[9]  = (_Float16)a2.y;
            af[10] = (_Float16)a2.z; af[11] = (_Float16)a2.w;
            af[12] = (_Float16)a3.x; af[13] = (_Float16)a3.y;
            af[14] = (_Float16)a3.z; af[15] = (_Float16)a3.w;
        } else {
            // branch-free masked tail: clamp address, select value
#pragma unroll
            for (int j = 0; j < 8; ++j) {
                const int ka = kaBase + j;
                const int kb = kaBase + 16 + j;
                const float xa = Arow[ka < K ? ka : 0];
                const float xb = Arow[kb < K ? kb : 0];
                af[j]     = (_Float16)(ka < K ? xa : 0.f);
                af[8 + j] = (_Float16)(kb < K ? xb : 0.f);
            }
        }
#pragma unroll
        for (int j = 0; j < 8; ++j) {
            bf[j]     = (_Float16)Bs[nn * BSTR + 8 * hi + j];
            bf[8 + j] = (_Float16)Bs[nn * BSTR + 16 + 8 * hi + j];
        }

        acc = __builtin_amdgcn_wmma_f32_16x16x32_f16(
                  false, af, false, bf, (short)0, acc, false, false);
        __syncthreads();
    }

    const float bv = bias ? bias[n] : 0.f;
#pragma unroll
    for (int j = 0; j < 8; ++j) {
        const int row = row0 + j + 8 * hi;
        float v = acc[j] + bv;
        if (act) v = eluf(v);
        C[(size_t)row * Nout + n] = v;
    }
}

// ---------------------------------------------------------------------------
// PyG LayerNorm(mode='graph') over all 32 nodes x 128 ch of each graph,
// optional ELU. One 256-thread workgroup per graph.
// ---------------------------------------------------------------------------
__global__ void __launch_bounds__(256)
k_ln_graph(const float* __restrict__ X, float* __restrict__ Y,
           const float* __restrict__ w, const float* __restrict__ b, int do_elu)
{
    __shared__ float red[256];
    const int g = blockIdx.x, t = threadIdx.x;
    const size_t base = (size_t)g * NPGc * HIDc;
    float vals[16];
    float s = 0.f, s2 = 0.f;
#pragma unroll
    for (int i = 0; i < 16; ++i) {
        float x = X[base + t + 256 * i];
        vals[i] = x; s += x; s2 += x * x;
    }
    red[t] = s; __syncthreads();
    for (int o = 128; o > 0; o >>= 1) { if (t < o) red[t] += red[t + o]; __syncthreads(); }
    const float mean = red[0] * (1.f / 4096.f);
    __syncthreads();
    red[t] = s2; __syncthreads();
    for (int o = 128; o > 0; o >>= 1) { if (t < o) red[t] += red[t + o]; __syncthreads(); }
    const float var = red[0] * (1.f / 4096.f) - mean * mean;
    const float rs  = rsqrtf(var + 1e-5f);
#pragma unroll
    for (int i = 0; i < 16; ++i) {
        const int idx = t + 256 * i;
        const int c   = idx & (HIDc - 1);
        float y = (vals[i] - mean) * rs * w[c] + b[c];
        if (do_elu) y = eluf(y);
        Y[base + idx] = y;
    }
}

// ---------------------------------------------------------------------------
// TransformerConv gather/softmax/scatter for one graph (32 nodes, 64 edges).
// Q,K,V: [N,128] (head-major h*64+c), Eproj: [E,128].
// Y is pre-filled with the root/skip term (x@Ws+bs); attention sum is added.
// ---------------------------------------------------------------------------
__global__ void __launch_bounds__(256)
k_tconv(const float* __restrict__ Q, const float* __restrict__ Kf,
        const float* __restrict__ V, const float* __restrict__ Eproj,
        const int* __restrict__ src, const int* __restrict__ dst,
        float* __restrict__ Y)
{
    __shared__ float qs[NPGc * HIDc], ks[NPGc * HIDc], vs[NPGc * HIDc];
    __shared__ float aw[EPGc * 2];
    __shared__ float mx[NPGc * 2], sden[NPGc * 2];
    __shared__ int   sl[EPGc], dl[EPGc];

    const int g = blockIdx.x, t = threadIdx.x;
    const size_t nb    = (size_t)g * NPGc * HIDc;
    const size_t ebase = (size_t)g * EPGc * HIDc;

    for (int i = t; i < NPGc * HIDc; i += 256) {
        qs[i] = Q[nb + i]; ks[i] = Kf[nb + i]; vs[i] = V[nb + i];
    }
    if (t < EPGc) {
        sl[t] = src[g * EPGc + t] - g * NPGc;
        dl[t] = dst[g * EPGc + t] - g * NPGc;
    }
    __syncthreads();

    // scores per (edge, head)
    if (t < EPGc * 2) {
        const int e = t >> 1, h = t & 1;
        const float* ep = &Eproj[ebase + (size_t)e * HIDc + h * 64];
        const float* kp = &ks[sl[e] * HIDc + h * 64];
        const float* qp = &qs[dl[e] * HIDc + h * 64];
        float a = 0.f;
        for (int c = 0; c < 64; ++c) a += qp[c] * (kp[c] + ep[c]);
        aw[t] = a * 0.125f;             // 64^{-1/2}
    }
    __syncthreads();

    // per-(dst,head) max & exp-sum
    if (t < NPGc * 2) {
        const int n = t >> 1, h = t & 1;
        float m_ = -3.0e38f;
        for (int e = 0; e < EPGc; ++e)
            if (dl[e] == n) { float a = aw[e * 2 + h]; m_ = a > m_ ? a : m_; }
        float s_ = 0.f;
        for (int e = 0; e < EPGc; ++e)
            if (dl[e] == n) s_ += __expf(aw[e * 2 + h] - m_);
        mx[t] = m_; sden[t] = s_;
    }
    __syncthreads();
    if (t < EPGc * 2) {
        const int e = t >> 1, h = t & 1, n = dl[e];
        aw[t] = __expf(aw[t] - mx[n * 2 + h]) / (sden[n * 2 + h] + 1e-16f);
    }
    __syncthreads();

    // scatter: 32 nodes x 128 ch
    for (int it = 0; it < 16; ++it) {
        const int idx = t + 256 * it;
        const int n = idx >> 7, ch = idx & 127, h = ch >> 6;
        float acc = 0.f;
        for (int e = 0; e < EPGc; ++e)
            if (dl[e] == n)
                acc += (vs[sl[e] * HIDc + ch] + Eproj[ebase + (size_t)e * HIDc + ch])
                       * aw[e * 2 + h];
        Y[nb + idx] += acc;
    }
}

// ---------------------------------------------------------------------------
// SAGPooling(min_score=-1) + score-weighted add-pool + edge-merge + L2 norm.
// One 128-thread workgroup per graph; writes OUT[g, blk, 0:128].
// ---------------------------------------------------------------------------
__global__ void __launch_bounds__(128)
k_pool(const float* __restrict__ X, const float* __restrict__ EA,
       const int* __restrict__ src, const int* __restrict__ dst,
       const float* __restrict__ w_rel, const float* __restrict__ b_rel,
       const float* __restrict__ w_root,
       const float* __restrict__ W_re, const float* __restrict__ b_re,
       float* __restrict__ OUT, int blk)
{
    __shared__ float xs[NPGc * HIDc];
    __shared__ float p[NPGc], pr[NPGc], sc[NPGc];
    __shared__ float easum[64];
    __shared__ float gv[HIDc];
    __shared__ float red[128];
    __shared__ int   sl[EPGc], dl[EPGc];

    const int g = blockIdx.x, t = threadIdx.x;
    const size_t nb = (size_t)g * NPGc * HIDc;
    for (int i = t; i < NPGc * HIDc; i += 128) xs[i] = X[nb + i];
    if (t < EPGc) {
        sl[t] = src[g * EPGc + t] - g * NPGc;
        dl[t] = dst[g * EPGc + t] - g * NPGc;
    }
    __syncthreads();

    if (t < NPGc) {
        float a = 0.f, r = 0.f;
        for (int c = 0; c < HIDc; ++c) {
            a += xs[t * HIDc + c] * w_rel[c];
            r += xs[t * HIDc + c] * w_root[c];
        }
        p[t] = a; pr[t] = r;
    }
    __syncthreads();
    if (t < NPGc) {
        float s_ = b_rel[0] + pr[t];
        for (int e = 0; e < EPGc; ++e) if (dl[e] == t) s_ += p[sl[e]];
        sc[t] = s_;
    }
    __syncthreads();
    if (t == 0) {
        float m_ = -3.0e38f;
        for (int n = 0; n < NPGc; ++n) m_ = sc[n] > m_ ? sc[n] : m_;
        float s_ = 0.f;
        for (int n = 0; n < NPGc; ++n) { sc[n] = __expf(sc[n] - m_); s_ += sc[n]; }
        const float inv = 1.f / (s_ + 1e-16f);
        for (int n = 0; n < NPGc; ++n) sc[n] *= inv;
    }
    __syncthreads();

    { // weighted pool
        float a = 0.f;
        for (int n = 0; n < NPGc; ++n) a += sc[n] * xs[n * HIDc + t];
        gv[t] = a;
    }
    if (t < 64) {
        float a = 0.f;
        const float* eb = &EA[(size_t)g * EPGc * 64];
        for (int e = 0; e < EPGc; ++e) a += eb[e * 64 + t];
        easum[t] = a;
    }
    __syncthreads();

    float a = b_re[t];
    for (int kk = 0; kk < 64; ++kk) a += easum[kk] * W_re[kk * HIDc + t];
    const float gval = gv[t] * eluf(a);

    red[t] = gval * gval; __syncthreads();
    for (int o = 64; o > 0; o >>= 1) { if (t < o) red[t] += red[t + o]; __syncthreads(); }
    const float nrm = fmaxf(sqrtf(red[0]), 1e-12f);
    OUT[(size_t)g * (NBLKc * HIDc) + blk * HIDc + t] = gval / nrm;
}

// ---------------------------------------------------------------------------
// Plain per-row LayerNorm over 128 channels (co-attention pre-LN).
// ---------------------------------------------------------------------------
__global__ void __launch_bounds__(128)
k_ln_rows(const float* __restrict__ X, float* __restrict__ Y,
          const float* __restrict__ w, const float* __restrict__ b)
{
    __shared__ float red[128];
    const int r = blockIdx.x, t = threadIdx.x;
    const float x = X[(size_t)r * HIDc + t];
    red[t] = x; __syncthreads();
    for (int o = 64; o > 0; o >>= 1) { if (t < o) red[t] += red[t + o]; __syncthreads(); }
    const float mean = red[0] * (1.f / HIDc);
    __syncthreads();
    const float d = x - mean;
    red[t] = d * d; __syncthreads();
    for (int o = 64; o > 0; o >>= 1) { if (t < o) red[t] += red[t + o]; __syncthreads(); }
    const float var = red[0] * (1.f / HIDc);
    Y[(size_t)r * HIDc + t] = d * rsqrtf(var + 1e-5f) * w[t] + b[t];
}

// ---------------------------------------------------------------------------
// Co-attention core: 4 tokens, 8 heads, 128 dim. One block per (b,h).
// Qb: [B,4,1024], KVb: [B,4,2048] (k | v), Ob: [B,4,1024].
// ---------------------------------------------------------------------------
__global__ void __launch_bounds__(128)
k_attn(const float* __restrict__ Qb, const float* __restrict__ KVb,
       float* __restrict__ Ob)
{
    __shared__ float qsm[4 * 128], ksm[4 * 128], vsm[4 * 128];
    __shared__ float s[16];
    const int bh = blockIdx.x, b = bh >> 3, h = bh & 7, t = threadIdx.x;
    for (int n = 0; n < 4; ++n) {
        qsm[n * 128 + t] = Qb[((size_t)b * 4 + n) * 1024 + h * 128 + t];
        ksm[n * 128 + t] = KVb[((size_t)b * 4 + n) * 2048 + h * 128 + t];
        vsm[n * 128 + t] = KVb[((size_t)b * 4 + n) * 2048 + 1024 + h * 128 + t];
    }
    __syncthreads();
    if (t < 16) {
        const int n = t >> 2, m = t & 3;
        float a = 0.f;
        for (int d = 0; d < 128; ++d) a += qsm[n * 128 + d] * ksm[m * 128 + d];
        s[t] = a * 0.08838834764831845f;   // 128^{-1/2}
    }
    __syncthreads();
    if (t < 4) {
        const float m_ = fmaxf(fmaxf(s[t*4], s[t*4+1]), fmaxf(s[t*4+2], s[t*4+3]));
        float e0 = __expf(s[t*4]   - m_), e1 = __expf(s[t*4+1] - m_);
        float e2 = __expf(s[t*4+2] - m_), e3 = __expf(s[t*4+3] - m_);
        const float inv = 1.f / (e0 + e1 + e2 + e3);
        s[t*4] = e0*inv; s[t*4+1] = e1*inv; s[t*4+2] = e2*inv; s[t*4+3] = e3*inv;
    }
    __syncthreads();
    for (int n = 0; n < 4; ++n) {
        const float o = s[n*4]*vsm[t] + s[n*4+1]*vsm[128+t]
                      + s[n*4+2]*vsm[256+t] + s[n*4+3]*vsm[384+t];
        Ob[((size_t)b * 4 + n) * 1024 + h * 128 + t] = o;
    }
}

// ---------------------------------------------------------------------------
// Final bilinear merge + classifier softmax. One wave per graph-pair.
// ---------------------------------------------------------------------------
__global__ void __launch_bounds__(32)
k_final(const float* __restrict__ HA, const float* __restrict__ TA,
        const float* __restrict__ Wm, const float* __restrict__ bm,
        float* __restrict__ OUT)
{
    __shared__ float m16[16];
    const int b = blockIdx.x, t = threadIdx.x;
    if (t < 16) {
        const int n = t >> 2, m = t & 3;
        float a = 0.f;
        for (int d = 0; d < 128; ++d)
            a += HA[((size_t)b * 4 + n) * 128 + d] * TA[((size_t)b * 4 + m) * 128 + d];
        m16[t] = a;
    }
    __syncthreads();
    if (t == 0) {
        float l0 = bm[0], l1 = bm[1];
        for (int p = 0; p < 16; ++p) { l0 += m16[p] * Wm[p * 2]; l1 += m16[p] * Wm[p * 2 + 1]; }
        const float mx_ = fmaxf(l0, l1);
        const float e0 = __expf(l0 - mx_), e1 = __expf(l1 - mx_);
        const float inv = 1.f / (e0 + e1);
        OUT[(size_t)b * 2]     = e0 * inv;
        OUT[(size_t)b * 2 + 1] = e1 * inv;
    }
}

// ---------------------------------------------------------------------------
// Host orchestration
// ---------------------------------------------------------------------------
extern "C" void kernel_launch(void* const* d_in, const int* in_sizes, int n_in,
                              void* d_out, int out_size, void* d_ws, size_t ws_size,
                              hipStream_t stream)
{
    (void)in_sizes; (void)n_in; (void)out_size; (void)ws_size;

    const float* x1     = (const float*)d_in[0];
    const float* x2     = (const float*)d_in[1];
    const float* ea1    = (const float*)d_in[2];
    const float* ea2    = (const float*)d_in[3];
    const float* W_node = (const float*)d_in[4];
    const float* b_node = (const float*)d_in[5];
    const float* W_edge = (const float*)d_in[6];
    const float* b_edge = (const float*)d_in[7];
    const float* ln0_w  = (const float*)d_in[8];
    const float* ln0_b  = (const float*)d_in[9];
    const float* Wq     = (const float*)d_in[10];
    const float* bq     = (const float*)d_in[11];
    const float* Wk     = (const float*)d_in[12];
    const float* bk     = (const float*)d_in[13];
    const float* Wv     = (const float*)d_in[14];
    const float* bv     = (const float*)d_in[15];
    const float* We     = (const float*)d_in[16];
    const float* Ws     = (const float*)d_in[17];
    const float* bs     = (const float*)d_in[18];
    const float* Wup    = (const float*)d_in[19];
    const float* bup    = (const float*)d_in[20];
    const float* lnw    = (const float*)d_in[21];
    const float* lnb    = (const float*)d_in[22];
    const float* w_rel  = (const float*)d_in[23];
    const float* b_rel  = (const float*)d_in[24];
    const float* w_root = (const float*)d_in[25];
    const float* W_re   = (const float*)d_in[26];
    const float* b_re   = (const float*)d_in[27];
    const float* ca_lw  = (const float*)d_in[28];
    const float* ca_lb  = (const float*)d_in[29];
    const float* Wq_ca  = (const float*)d_in[30];
    const float* Wkv_ca = (const float*)d_in[31];
    const float* Wo_ca  = (const float*)d_in[32];
    const float* W_mrg  = (const float*)d_in[33];
    const float* b_mrg  = (const float*)d_in[34];
    const int*   ei1    = (const int*)d_in[35];
    const int*   ei2    = (const int*)d_in[36];
    // d_in[37..40] batch/ebatch are implicit (block-contiguous) -> unused.

    // ---- workspace layout (bytes) ----
    const size_t NB = (size_t)NTOT * HIDc * sizeof(float);     // 32 MiB
    char* ws = (char*)d_ws;
    float* bx   = (float*)(ws + 0 * NB);       // current node feats [N,128]
    float* bxo  = (float*)(ws + 1 * NB);       // conv output / skip  [N,128]
    float* bqb  = (float*)(ws + 2 * NB);       // q [N,128]
    float* bkb  = (float*)(ws + 3 * NB);       // k [N,128]
    float* bvb  = (float*)(ws + 4 * NB);       // v [N,128]
    float* beb  = (float*)(ws + 5 * NB);       // edge proj [E,128] (2*NB)
    float* bea  = (float*)(ws + 7 * NB);       // ea ping  [E,64]
    float* beat = (float*)(ws + 8 * NB);       // ea pong  [E,64]
    float* rh   = (float*)(ws + 9 * NB);                         // [B,4,128]
    float* rt   = (float*)(ws + 9 * NB + (size_t)4 * 1024 * 1024);
    // co-attention overlays drug region [0, 9*NB)
    float* lnH = (float*)(ws + 0);
    float* lnT = (float*)(ws + (size_t)4  * 1024 * 1024);
    float* qbf = (float*)(ws + (size_t)8  * 1024 * 1024);        // [8192,1024]
    float* kvb = (float*)(ws + (size_t)40 * 1024 * 1024);        // [8192,2048]
    float* obf = (float*)(ws + (size_t)104 * 1024 * 1024);       // [8192,1024]
    float* haB = (float*)(ws + (size_t)136 * 1024 * 1024);       // [8192,128]
    float* taB = (float*)(ws + (size_t)140 * 1024 * 1024);       // [8192,128]

    auto gemm = [&](const float* A, const float* Wt, const float* bias, float* C,
                    int M, int K, int Nout, int act) {
        dim3 gr(M / 64, Nout / 16);
        switch (K) {
        case EINc: k_gemm<EINc><<<gr, dim3(128), 0, stream>>>(A, Wt, bias, C, Nout, act); break;
        case 64:   k_gemm<64>  <<<gr, dim3(128), 0, stream>>>(A, Wt, bias, C, Nout, act); break;
        case FINc: k_gemm<FINc><<<gr, dim3(128), 0, stream>>>(A, Wt, bias, C, Nout, act); break;
        case HIDc: k_gemm<HIDc><<<gr, dim3(128), 0, stream>>>(A, Wt, bias, C, Nout, act); break;
        case 1024: k_gemm<1024><<<gr, dim3(128), 0, stream>>>(A, Wt, bias, C, Nout, act); break;
        default:   break;
        }
    };

    for (int drug = 0; drug < 2; ++drug) {
        const float* xin  = drug ? x2 : x1;
        const float* eain = drug ? ea2 : ea1;
        const int*   src  = drug ? ei2 : ei1;
        const int*   dst  = src + ETOT;
        float*       RH   = drug ? rt : rh;

        // init embeddings
        gemm(xin, W_node, b_node, bxo, NTOT, FINc, HIDc, 0);
        k_ln_graph<<<BGR, 256, 0, stream>>>(bxo, bx, ln0_w, ln0_b, 1);
        gemm(eain, W_edge, b_edge, bea, ETOT, EINc, 64, 1);

        float* eaA = bea;
        float* eaB = beat;
        for (int i = 0; i < NBLKc; ++i) {
            for (int ph = 0; ph < 2; ++ph) {
                const size_t li = (size_t)(i * 2 + ph);
                const float* Wq_ = Wq + li * HIDc * HIDc;  const float* bq_ = bq + li * HIDc;
                const float* Wk_ = Wk + li * HIDc * HIDc;  const float* bk_ = bk + li * HIDc;
                const float* Wv_ = Wv + li * HIDc * HIDc;  const float* bv_ = bv + li * HIDc;
                const float* We_ = We + li * 64 * HIDc;
                const float* Ws_ = Ws + li * HIDc * HIDc;  const float* bs_ = bs + li * HIDc;
                const float* Wu_ = Wup + li * 64 * 64;     const float* bu_ = bup + li * 64;
                const float* lw_ = lnw + li * HIDc;        const float* lb_ = lnb + li * HIDc;

                gemm(bx, Wq_, bq_, bqb, NTOT, HIDc, HIDc, 0);
                gemm(bx, Wk_, bk_, bkb, NTOT, HIDc, HIDc, 0);
                gemm(bx, Wv_, bv_, bvb, NTOT, HIDc, HIDc, 0);
                gemm(bx, Ws_, bs_, bxo, NTOT, HIDc, HIDc, 0);   // skip term
                gemm(eaA, We_, nullptr, beb, ETOT, 64, HIDc, 0);
                k_tconv<<<BGR, 256, 0, stream>>>(bqb, bkb, bvb, beb, src, dst, bxo);
                gemm(eaA, Wu_, bu_, eaB, ETOT, 64, 64, 1);      // edge update
                k_ln_graph<<<BGR, 256, 0, stream>>>(bxo, bx, lw_, lb_, 1);
                { float* tmp = eaA; eaA = eaB; eaB = tmp; }
            }
            k_pool<<<BGR, 128, 0, stream>>>(bx, eaA, src, dst, w_rel, b_rel,
                                            w_root, W_re, b_re, RH, i);
        }
    }

    // ---- co-attention over [B,4,128] summaries ----
    k_ln_rows<<<BGR * 4, 128, 0, stream>>>(rh, lnH, ca_lw, ca_lb);
    k_ln_rows<<<BGR * 4, 128, 0, stream>>>(rt, lnT, ca_lw, ca_lb);

    // ha = coattn(q from rh, kv from rt)
    gemm(lnH, Wq_ca, nullptr, qbf, BGR * 4, HIDc, CAHc * CADc, 0);
    gemm(lnT, Wkv_ca, nullptr, kvb, BGR * 4, HIDc, 2 * CAHc * CADc, 0);
    k_attn<<<BGR * CAHc, 128, 0, stream>>>(qbf, kvb, obf);
    gemm(obf, Wo_ca, nullptr, haB, BGR * 4, CAHc * CADc, HIDc, 0);

    // ta = coattn(q from rt, kv from rh)
    gemm(lnT, Wq_ca, nullptr, qbf, BGR * 4, HIDc, CAHc * CADc, 0);
    gemm(lnH, Wkv_ca, nullptr, kvb, BGR * 4, HIDc, 2 * CAHc * CADc, 0);
    k_attn<<<BGR * CAHc, 128, 0, stream>>>(qbf, kvb, obf);
    gemm(obf, Wo_ca, nullptr, taB, BGR * 4, CAHc * CADc, HIDc, 0);

    k_final<<<BGR, 32, 0, stream>>>(haB, taB, W_mrg, b_mrg, (float*)d_out);
}
